// MultiTimeAttention_12747462934580
// MI455X (gfx1250) — matrix operand order repaired and checked by reading, not compiled, not run on hardware
//
#include <hip/hip_runtime.h>

// ---------------------------------------------------------------------------
// MultiTimeAttention for MI455X (gfx1250, wave32, WMMA 16x16x32 f16)
// B=8 LQ=64 LK=256 E=128 H=4 EK=32 D=32 NH=128
// - All LDS operands stored so fragment gathers are contiguous-per-lane
//   (2x ds_load_b128) with padded strides (dword stride == 4 mod 64 -> max
//   2-way conflict on the 64-bank LDS).
// - Softmax row-max + exp fused in-register on the score fragments; score
//   fragments are kept packed-f16 (4 VGPRs each) to stay spill-free.
// ---------------------------------------------------------------------------

typedef __attribute__((ext_vector_type(16))) _Float16 v16h;
typedef __attribute__((ext_vector_type(8)))  _Float16 v8h;
typedef __attribute__((ext_vector_type(8)))  float    v8f;

#define SX_LD 136   // 64x128 slab, padded
#define SW_LD 136   // 128x128 W^T, padded
#define SQ_LD 40    // 64x32 q, padded
#define SK_LD 40    // 256x32 k, padded
#define SB_LD 264   // 64x256 [mask*v | mask]^T, padded
#define SE_LD 264   // 64x256 e, padded

__device__ __forceinline__ v8f wmma_f16(v16h a, v16h b, v8f c) {
  return __builtin_amdgcn_wmma_f32_16x16x32_f16(
      /*neg_a=*/false, a, /*neg_b=*/false, b,
      /*c_mod=*/(short)0, c, /*reuse_a=*/false, /*reuse_b=*/false);
}

// A fragment: 16x32 tile (M x K) of a row-major f16 matrix in LDS.
// lane L -> M = L%16, g = L/16; half pair v: K = 2*(v%4)+p + 8g + 16*(v/4)
// => per lane: halves 0..7 K-contiguous, halves 8..15 K-contiguous (2xb128)
__device__ __forceinline__ v16h load_frag_a(const _Float16* s, int mr, int kb, int ld) {
  const int lane = threadIdx.x & 31;
  const int m = mr + (lane & 15);
  const int g = lane >> 4;
  v16h a;
#pragma unroll
  for (int e = 0; e < 16; ++e) {
    const int v = e >> 1, p = e & 1;
    const int k = kb + ((v & 3) << 1) + p + (g << 3) + ((v >> 2) << 4);
    a[e] = s[m * ld + k];
  }
  return a;
}

// B fragment (32x16, K x N) taken as the TRANSPOSE of a row-major matrix T
// where T[n][k] = B[k][n].  lane L -> N = L%16, g = L/16; half e -> K = (e%8)+8g+16*(e/8)
// => per lane: halves 0..7 K-contiguous, halves 8..15 K-contiguous (2xb128).
__device__ __forceinline__ v16h load_frag_bt(const _Float16* s, int kb, int nc, int ld) {
  const int lane = threadIdx.x & 31;
  const int n = nc + (lane & 15);
  const int g = lane >> 4;
  v16h b;
#pragma unroll
  for (int e = 0; e < 16; ++e) {
    const int k = kb + (e & 7) + (g << 3) + ((e >> 3) << 4);
    b[e] = s[n * ld + k];
  }
  return b;
}

// ---------------------------------------------------------------------------
// Projection: Y = (X @ W + bias) * scale
// X: rows x 128 (f32 or f16), W: 128x128 f32, Y: rows x 128 (f16 or f32)
// grid.x = rows/64, block = 128 (4 waves); wave w owns M-band 16w.
// ---------------------------------------------------------------------------
template <bool IN_F16, bool OUT_F32>
__global__ void __launch_bounds__(128, 1)
proj_kernel(const void* __restrict__ Xv,
            const float* __restrict__ W,
            const float* __restrict__ bias,
            void* __restrict__ Yv, float scale) {
  __shared__ _Float16 sx[64 * SX_LD];
  __shared__ _Float16 swT[128 * SW_LD];
  __shared__ float    sb[128];

  const int t = threadIdx.x;
  const int row0 = blockIdx.x * 64;

  // Start pulling W toward L2 (global_prefetch_b8).
  __builtin_prefetch((const void*)(W + (t << 5)), 0, 1);

  if (IN_F16) {
    const _Float16* X = (const _Float16*)Xv;
    for (int i = t; i < 64 * 128; i += 128)
      sx[(i >> 7) * SX_LD + (i & 127)] = X[(row0 + (i >> 7)) * 128 + (i & 127)];
  } else {
    const float* X = (const float*)Xv;
    for (int i = t; i < 64 * 128; i += 128)
      sx[(i >> 7) * SX_LD + (i & 127)] = (_Float16)X[(row0 + (i >> 7)) * 128 + (i & 127)];
  }
  // Transposed W staging: global reads coalesced (n fastest); LDS write
  // stride 272B = 68 dwords == 4 (mod 64) -> 2-way conflict worst case.
  for (int i = t; i < 128 * 128; i += 128) {
    const int k = i >> 7, n = i & 127;
    swT[n * SW_LD + k] = (_Float16)W[i];
  }
  if (t < 128) sb[t] = bias[t];
  __syncthreads();

  const int w  = t >> 5;
  const int mr = w * 16;
  const int lane = t & 31;
  const int n = lane & 15;
  const int g = lane >> 4;

  // Hoist the 4 A fragments (K = 0,32,64,96), reused across all 8 N tiles.
  v16h a0 = load_frag_a(sx, mr, 0,  SX_LD);
  v16h a1 = load_frag_a(sx, mr, 32, SX_LD);
  v16h a2 = load_frag_a(sx, mr, 64, SX_LD);
  v16h a3 = load_frag_a(sx, mr, 96, SX_LD);

#pragma unroll
  for (int nt = 0; nt < 8; ++nt) {
    v8f c = {};
    c = wmma_f16(a0, load_frag_bt(swT, 0,  nt * 16, SW_LD), c);
    c = wmma_f16(a1, load_frag_bt(swT, 32, nt * 16, SW_LD), c);
    c = wmma_f16(a2, load_frag_bt(swT, 64, nt * 16, SW_LD), c);
    c = wmma_f16(a3, load_frag_bt(swT, 96, nt * 16, SW_LD), c);
    const int col = nt * 16 + n;
    const float bb = sb[col];
    if (OUT_F32) {
      float* Y = (float*)Yv;
#pragma unroll
      for (int r = 0; r < 8; ++r)
        Y[(row0 + mr + r + g * 8) * 128 + col] = (c[r] + bb) * scale;
    } else {
      _Float16* Y = (_Float16*)Yv;
#pragma unroll
      for (int r = 0; r < 8; ++r)
        Y[(row0 + mr + r + g * 8) * 128 + col] = (_Float16)((c[r] + bb) * scale);
    }
  }
}

// ---------------------------------------------------------------------------
// Attention per (b,h): scores (in reg, packed f16) -> fused rowmax+exp
//                      -> e to LDS -> num/den GEMMs -> x = num/den
// grid.x = B*H = 32, block = 128 (4 waves)
// ---------------------------------------------------------------------------
__global__ void __launch_bounds__(128, 1)
attn_kernel(const _Float16* __restrict__ qg,   // 512 x 128 (scaled)
            const _Float16* __restrict__ kg,   // 2048 x 128
            const float* __restrict__ value,   // B x LK x D
            const int* __restrict__ mask,      // B x LK x D
            _Float16* __restrict__ xg) {       // 512 x 128
  __shared__ _Float16 sq[64 * SQ_LD];      //  ~5 KB
  __shared__ _Float16 sk[256 * SK_LD];     // ~20 KB
  __shared__ _Float16 sbT[64 * SB_LD];     // ~33 KB : [mask*v | mask]^T
  __shared__ _Float16 se[64 * SE_LD];      // ~33 KB : e = exp(s - rowmax)

  const int t = threadIdx.x;
  const int b = blockIdx.x >> 2;
  const int h = blockIdx.x & 3;

  __builtin_prefetch((const void*)(value + (b * 256) * 32 + (t << 6)), 0, 1);

  for (int i = t; i < 64 * 32; i += 128) {
    const int r = i >> 5, c = i & 31;
    sq[r * SQ_LD + c] = qg[(b * 64 + r) * 128 + h * 32 + c];
  }
  for (int i = t; i < 256 * 32; i += 128) {
    const int r = i >> 5, c = i & 31;
    sk[r * SK_LD + c] = kg[(b * 256 + r) * 128 + h * 32 + c];
  }
  // Transposed B staging for num/den: rows 0..31 = mask*v, rows 32..63 = mask.
  // Global reads coalesced (d fastest per wave); LDS write stride 528B = 132
  // dwords == 4 (mod 64) -> 2-way conflict worst case.
  for (int i = t; i < 256 * 32; i += 128) {
    const int lk = i >> 5, d = i & 31;
    const int m  = mask[(b * 256 + lk) * 32 + d];
    const float v = value[(b * 256 + lk) * 32 + d];
    sbT[d * SB_LD + lk]        = m ? (_Float16)v : (_Float16)0.0f;
    sbT[(32 + d) * SB_LD + lk] = (_Float16)(float)m;
  }
  __syncthreads();

  const int w  = t >> 5;
  const int mr = w * 16;
  const int lane = t & 31;
  const int n = lane & 15;
  const int g = lane >> 4;

  // ---- scores for the wave's full 16-row band, kept packed f16 ------------
  // D-fragment layout: lane -> column n (+16*nt), VGPR r -> row mr + r + 8g.
  {
    v16h a = load_frag_a(sq, mr, 0, SQ_LD);   // single K step (K = 32)
    v8h sch[16];                              // 16 x 4 VGPRs = 64 VGPRs
    float m[8];
#pragma unroll
    for (int r = 0; r < 8; ++r) m[r] = -1e30f;

#pragma unroll
    for (int nt = 0; nt < 16; ++nt) {
      v8f c = {};
      c = wmma_f16(a, load_frag_bt(sk, 0, nt * 16, SK_LD), c);
      v8h ch;
#pragma unroll
      for (int r = 0; r < 8; ++r) {
        m[r] = fmaxf(m[r], c[r]);
        ch[r] = (_Float16)c[r];
      }
      sch[nt] = ch;
    }

    // Cross-lane row max over the 16 lanes of this half-wave
    // (width-16 xor shuffles; rows stay within their half).
#pragma unroll
    for (int r = 0; r < 8; ++r) {
#pragma unroll
      for (int s = 1; s < 16; s <<= 1) m[r] = fmaxf(m[r], __shfl_xor(m[r], s, 16));
    }

    // e = exp(s - rowmax); single LDS write pass (column-striped b16 stores).
#pragma unroll
    for (int nt = 0; nt < 16; ++nt) {
#pragma unroll
      for (int r = 0; r < 8; ++r) {
        se[(mr + r + g * 8) * SE_LD + nt * 16 + n] =
            (_Float16)__expf((float)sch[nt][r] - m[r]);
      }
    }
  }
  __syncthreads();

  // ---- num = e @ (mask*v), den = e @ mask ; x = num/den -------------------
  {
    // Hoist all 8 A fragments of the K=256 loop (64 VGPRs).
    v16h a[8];
#pragma unroll
    for (int i = 0; i < 8; ++i) a[i] = load_frag_a(se, mr, i * 32, SE_LD);

    v8f cn[2], cd[2];
#pragma unroll
    for (int nt = 0; nt < 2; ++nt) {
      v8f c0 = {}, c1 = {};
#pragma unroll
      for (int i = 0; i < 8; ++i) {
        c0 = wmma_f16(a[i], load_frag_bt(sbT, i * 32, nt * 16, SB_LD), c0);       // num
        c1 = wmma_f16(a[i], load_frag_bt(sbT, i * 32, 32 + nt * 16, SB_LD), c1);  // den
      }
      cn[nt] = c0; cd[nt] = c1;
    }
#pragma unroll
    for (int nt = 0; nt < 2; ++nt) {
#pragma unroll
      for (int r = 0; r < 8; ++r) {
        const int d = nt * 16 + n;
        const float x = cn[nt][r] / cd[nt][r];
        xg[(b * 64 + mr + r + g * 8) * 128 + h * 32 + d] = (_Float16)x;
      }
    }
  }
}

// ---------------------------------------------------------------------------
extern "C" void kernel_launch(void* const* d_in, const int* in_sizes, int n_in,
                              void* d_out, int out_size, void* d_ws, size_t ws_size,
                              hipStream_t stream) {
  const float* query = (const float*)d_in[0];
  const float* key   = (const float*)d_in[1];
  const float* value = (const float*)d_in[2];
  const int*   mask  = (const int*)d_in[3];
  const float* Wq    = (const float*)d_in[4];
  const float* bq    = (const float*)d_in[5];
  const float* Wk    = (const float*)d_in[6];
  const float* bk    = (const float*)d_in[7];
  const float* Wo    = (const float*)d_in[8];
  const float* bo    = (const float*)d_in[9];
  float* out = (float*)d_out;

  _Float16* qg = (_Float16*)d_ws;          // 512 x 128
  _Float16* kg = qg + 512 * 128;           // 2048 x 128
  _Float16* xg = kg + 2048 * 128;          // 512 x 128

  const float inv_sqrt_ek = 0.17677669529663687f;  // 1/sqrt(32), folded into q

  proj_kernel<false, false><<<8,  128, 0, stream>>>(query, Wq, bq, qg, inv_sqrt_ek);
  proj_kernel<false, false><<<32, 128, 0, stream>>>(key,   Wk, bk, kg, 1.0f);
  attn_kernel<<<32, 128, 0, stream>>>(qg, kg, value, mask, xg);
  proj_kernel<true,  true><<<8,  128, 0, stream>>>(xg, Wo, bo, out, 1.0f);
}
